// CAnet_22222160789565
// MI455X (gfx1250) — matrix operand order repaired and checked by reading, not compiled, not run on hardware
//
#include <hip/hip_runtime.h>
#include <stdint.h>

typedef __attribute__((ext_vector_type(16))) _Float16     v16h;
typedef __attribute__((ext_vector_type(8)))  float        v8f;
typedef __attribute__((ext_vector_type(4)))  unsigned int v4u;
typedef __attribute__((ext_vector_type(8)))  int          v8i;
typedef __attribute__((ext_vector_type(4)))  int          v4i;

#define HW3  14161          // 119*119
#define W3   119
#define NCOL 56644          // 4*119*119 (batch folded into n, as in reference)

#if defined(__has_builtin)
# if __has_builtin(__builtin_amdgcn_tensor_load_to_lds) && \
     __has_builtin(__builtin_amdgcn_s_wait_tensorcnt)
#  define HAVE_TDM 1
# endif
#endif

// ---------------------------------------------------------------------------
// Weight pack: w[64][Cin][3][3] fp32 -> wp[n][k] fp16 (n-major, row stride
// Kpad), k = cin*9+kh*3+kw, K zero-padded to a multiple of 32. The n-major
// layout makes each K-step's B tile (64 rows x 32 k) a dense 2D TDM tile.
// ---------------------------------------------------------------------------
__global__ __launch_bounds__(256) void pack_weights_k(const float* __restrict__ w,
    _Float16* __restrict__ wp, int Ktot, int Kpad) {
  int i = blockIdx.x * 256 + threadIdx.x;
  if (i >= Kpad * 64) return;
  int n = i / Kpad, k = i - n * Kpad;
  float v = (k < Ktot) ? w[n * Ktot + k] : 0.0f;
  wp[i] = (_Float16)v;
}

// ---------------------------------------------------------------------------
// Implicit-GEMM 3x3 (dilated) conv + bias + ReLU via v_wmma_f32_16x16x32_f16.
// GEMM view: M = B*Ho*Wo (pixels), N = 64 (Cout), K = Cin*9.
// Block: 256 threads = 8 waves; wave w -> (mh = w>>2, nt = w&3); each wave
// owns TWO 16x16 C tiles (rows mh*32..mh*32+31) sharing one B fragment.
// B tile is DMA'd into LDS by the Tensor Data Mover (wave 0 issues, waits on
// TENSORcnt, workgroup barrier publishes); A tile staged fp32->fp16 by VALU.
// ---------------------------------------------------------------------------
__global__ __launch_bounds__(256) void conv3x3_wmma_k(
    const float* __restrict__ in, const _Float16* __restrict__ wp,
    const float* __restrict__ bias, float* __restrict__ out,
    int B, int Cin, int Hin, int Win, int Ho, int Wo, int dil,
    int Ktot, int Ksteps) {
  __shared__ _Float16 Al[64][32];
  __shared__ _Float16 Bl[64][32];
  const int tid  = threadIdx.x;
  const int lane = tid & 31;
  const int wave = tid >> 5;
  const int mh   = wave >> 2;           // 0..1  -> rows mh*32 .. mh*32+31
  const int nt   = wave & 3;            // 0..3  -> channels nt*16 .. nt*16+15
  const int half = lane >> 4;
  const int l15  = lane & 15;
  const int HoWo = Ho * Wo;
  const int Mtot = B * HoWo;
  const int m0   = blockIdx.x * 64;
  const int Kpad = Ksteps * 32;

  v8f acc0 = {0.f, 0.f, 0.f, 0.f, 0.f, 0.f, 0.f, 0.f};
  v8f acc1 = {0.f, 0.f, 0.f, 0.f, 0.f, 0.f, 0.f, 0.f};

  for (int ks = 0; ks < Ksteps; ++ks) {
    const int k0 = ks * 32;
    // Stage A (64 rows x 32 k) fp32 -> fp16, gather from NCHW input.
    #pragma unroll
    for (int t = 0; t < 8; ++t) {
      int e  = tid + t * 256;
      int mi = e >> 5, kk = e & 31;
      int m = m0 + mi, k = k0 + kk;
      float v = 0.0f;
      if (m < Mtot && k < Ktot) {
        int bI = m / HoWo;
        int r  = m - bI * HoWo;
        int oh = r / Wo;
        int ow = r - oh * Wo;
        int ci = k / 9;
        int kr = k - ci * 9;
        int kh = kr / 3;
        int kw = kr - kh * 3;
        v = in[(((size_t)(bI * Cin + ci)) * Hin + (oh + kh * dil)) * Win
               + (ow + kw * dil)];
      }
      Al[mi][kk] = (_Float16)v;
    }
#ifdef HAVE_TDM
    // Tensor Data Mover: 2D tile (tile_dim0=32 k, tile_dim1=64 rows) of the
    // n-major weight matrix (tensor_dim0=Kpad, row stride Kpad) -> Bl[64][32].
    if (wave == 0) {
      unsigned long long ga =
          (unsigned long long)(uintptr_t)wp + (unsigned long long)k0 * 2ull;
      unsigned int ldsAddr = (unsigned int)(uintptr_t)(void*)&Bl[0][0];
      v4u g0;
      g0[0] = 1u;                                    // count=1 (valid), user D#
      g0[1] = ldsAddr;                               // lds_addr
      g0[2] = (unsigned int)(ga & 0xffffffffull);    // global_addr[31:0]
      g0[3] = (unsigned int)(((ga >> 32) & 0x01ffffffull) | 0x80000000u); // +type=2
      v8i g1;
      g1[0] = (int)(1u << 16);                       // data_size=1 (2 bytes)
      g1[1] = (int)((unsigned int)Kpad << 16);       // tensor_dim0[15:0]
      g1[2] = (int)((((unsigned int)Kpad >> 16) & 0xffffu) | (64u << 16)); // td0 hi | tensor_dim1=64
      g1[3] = (int)(32u << 16);                      // tile_dim0=32
      g1[4] = 64;                                    // tile_dim1=64, tile_dim2=0
      g1[5] = Kpad;                                  // tensor_dim0_stride[31:0]
      g1[6] = 0;
      g1[7] = 0;
      v4i g2 = {0, 0, 0, 0};
      v4i g3 = {0, 0, 0, 0};
#if __clang_major__ >= 23
      v8i g4 = {0, 0, 0, 0, 0, 0, 0, 0};
      __builtin_amdgcn_tensor_load_to_lds(g0, g1, g2, g3, g4, 0);
#else
      __builtin_amdgcn_tensor_load_to_lds(g0, g1, g2, g3, 0);
#endif
      __builtin_amdgcn_s_wait_tensorcnt(0);          // s_wait_tensorcnt 0
    }
#else
    // Fallback: manual coalesced staging of the B tile.
    #pragma unroll
    for (int t = 0; t < 8; ++t) {
      int e  = tid + t * 256;
      int ni = e >> 5, kk = e & 31;
      Bl[ni][kk] = wp[(size_t)ni * Kpad + (k0 + kk)];
    }
#endif
    if (k0 + 32 < Kpad)
      __builtin_prefetch(&wp[(size_t)(k0 + 32)], 0, 1);  // global_prefetch_b8
    __syncthreads();

    // A fragment per ISA 16-bit A 16x32 layout:
    //   a[i] = A[m][K_i], K_i = half*8 + (i&7) + (i&8)*2  (pairs contiguous)
    union { unsigned int u[8]; v16h h; } ua0, ua1, ub;
    const unsigned int* arow0 = (const unsigned int*)(&Al[mh * 32 + l15][0]);
    const unsigned int* arow1 = (const unsigned int*)(&Al[mh * 32 + 16 + l15][0]);
    #pragma unroll
    for (int q = 0; q < 4; ++q) {
      ua0.u[q]     = arow0[half * 4 + q];
      ua0.u[4 + q] = arow0[8 + half * 4 + q];
      ua1.u[q]     = arow1[half * 4 + q];
      ua1.u[4 + q] = arow1[8 + half * 4 + q];
    }
    // B fragment: lane holds column n = nt*16+l15, K_i = half*8 + (i&7) + (i&8)*2.
    const unsigned int* brow =
        (const unsigned int*)(&Bl[nt * 16 + l15][0]) + half * 4;
    #pragma unroll
    for (int q = 0; q < 8; ++q) ub.u[q] = brow[(q & 3) + (q >> 2) * 8];

    acc0 = __builtin_amdgcn_wmma_f32_16x16x32_f16(false, ua0.h, false, ub.h,
                                                  (short)0, acc0, false, false);
    acc1 = __builtin_amdgcn_wmma_f32_16x16x32_f16(false, ua1.h, false, ub.h,
                                                  (short)0, acc1, false, false);
    __syncthreads();
  }

  // C/D layout: lane -> N = l15; VGPR r -> M = r + 8*half.
  const int   nc = nt * 16 + l15;
  const float bb = bias[nc];
  #pragma unroll
  for (int r = 0; r < 8; ++r) {
    int m = m0 + mh * 32 + half * 8 + r;
    if (m < Mtot) {
      int bI  = m / HoWo;
      int rem = m - bI * HoWo;
      float v = acc0[r] + bb;
      out[((size_t)(bI * 64 + nc)) * HoWo + rem] = v > 0.f ? v : 0.f;
    }
    int m2 = m + 16;
    if (m2 < Mtot) {
      int bI  = m2 / HoWo;
      int rem = m2 - bI * HoWo;
      float v = acc1[r] + bb;
      out[((size_t)(bI * 64 + nc)) * HoWo + rem] = v > 0.f ? v : 0.f;
    }
  }
}

// ---------------------------------------------------------------------------
// 2x2 average pool, 250x250 -> 125x125
// ---------------------------------------------------------------------------
__global__ __launch_bounds__(256) void avgpool2_k(const float* __restrict__ c2,
                                                  float* __restrict__ p) {
  int i = blockIdx.x * 256 + threadIdx.x;
  const int tot = 4 * 64 * 125 * 125;
  if (i >= tot) return;
  int bc = i / (125 * 125);
  int r  = i - bc * (125 * 125);
  int y = r / 125, x = r - y * 125;
  const float* s = c2 + (size_t)bc * (250 * 250) + (2 * y) * 250 + 2 * x;
  p[i] = 0.25f * (s[0] + s[1] + s[250] + s[251]);
}

// ---------------------------------------------------------------------------
// Per (b,c): spatial mean & max of c3 (one block per channel).
// ---------------------------------------------------------------------------
__global__ __launch_bounds__(256) void chan_reduce_k(const float* __restrict__ c3,
    float* __restrict__ avg, float* __restrict__ mx) {
  __shared__ float ss[256], sm[256];
  int bc = blockIdx.x;
  const float* p = c3 + (size_t)bc * HW3;
  float s = 0.f, m = -3.0e38f;
  for (int i = threadIdx.x; i < HW3; i += 256) {
    float v = p[i]; s += v; m = fmaxf(m, v);
  }
  ss[threadIdx.x] = s; sm[threadIdx.x] = m;
  __syncthreads();
  for (int o = 128; o > 0; o >>= 1) {
    if (threadIdx.x < o) {
      ss[threadIdx.x] += ss[threadIdx.x + o];
      sm[threadIdx.x] = fmaxf(sm[threadIdx.x], sm[threadIdx.x + o]);
    }
    __syncthreads();
  }
  if (threadIdx.x == 0) { avg[bc] = ss[0] / (float)HW3; mx[bc] = sm[0]; }
}

// ---------------------------------------------------------------------------
// CBAM channel gate: gate[b,c] = sigmoid(fc(avg_b) + fc(max_b))[c]
// ---------------------------------------------------------------------------
__global__ __launch_bounds__(256) void gate_k(const float* __restrict__ avg,
    const float* __restrict__ mx, const float* __restrict__ fc1,
    const float* __restrict__ fc2, float* __restrict__ gate) {
  int tid = threadIdx.x;               // (b,c)
  int b = tid >> 6, c = tid & 63;
  float g = 0.f;
  for (int path = 0; path < 2; ++path) {
    const float* v = (path == 0) ? (avg + b * 64) : (mx + b * 64);
    #pragma unroll
    for (int j = 0; j < 4; ++j) {
      float h = 0.f;
      for (int cc = 0; cc < 64; ++cc) h += v[cc] * fc1[j * 64 + cc];
      h = h > 0.f ? h : 0.f;
      g += h * fc2[c * 4 + j];
    }
  }
  gate[tid] = 1.f / (1.f + __expf(-g));
}

// ---------------------------------------------------------------------------
// Spatial attention input: mean/max over channels of gate*c3 -> sp[b][2][hw]
// ---------------------------------------------------------------------------
__global__ __launch_bounds__(256) void spatial_pool_k(const float* __restrict__ c3,
    const float* __restrict__ gate, float* __restrict__ sp) {
  int i = blockIdx.x * 256 + threadIdx.x;
  if (i >= 4 * HW3) return;
  int b = i / HW3, p = i - b * HW3;
  float s = 0.f, m = -3.0e38f;
  for (int c = 0; c < 64; ++c) {
    float v = gate[b * 64 + c] * c3[((size_t)(b * 64 + c)) * HW3 + p];
    s += v; m = fmaxf(m, v);
  }
  sp[((size_t)(b * 2 + 0)) * HW3 + p] = s * (1.f / 64.f);
  sp[((size_t)(b * 2 + 1)) * HW3 + p] = m;
}

// ---------------------------------------------------------------------------
// 7x7 conv (2ch->1, pad 3) + sigmoid; xx = c3 * (1 + sig)  (c3 + sig*c3)
// ---------------------------------------------------------------------------
__global__ __launch_bounds__(256) void spatial_attn_k(const float* __restrict__ sp,
    const float* __restrict__ wp7, const float* __restrict__ c3,
    float* __restrict__ xx) {
  __shared__ float w7[98];
  if (threadIdx.x < 98) w7[threadIdx.x] = wp7[threadIdx.x];
  __syncthreads();
  int i = blockIdx.x * 256 + threadIdx.x;
  if (i >= 4 * HW3) return;
  int b = i / HW3, p = i - b * HW3;
  int y = p / W3, x = p - y * W3;
  float s = 0.f;
  for (int ch = 0; ch < 2; ++ch) {
    const float* sb = sp + ((size_t)(b * 2 + ch)) * HW3;
    for (int ky = 0; ky < 7; ++ky) {
      int iy = y + ky - 3;
      if (iy < 0 || iy >= W3) continue;
      for (int kx = 0; kx < 7; ++kx) {
        int ix = x + kx - 3;
        if (ix < 0 || ix >= W3) continue;
        s += w7[ch * 49 + ky * 7 + kx] * sb[iy * W3 + ix];
      }
    }
  }
  float f = 1.f + 1.f / (1.f + __expf(-s));
  for (int c = 0; c < 64; ++c) {
    size_t idx = ((size_t)(b * 64 + c)) * HW3 + p;
    xx[idx] = c3[idx] * f;
  }
}

// ---------------------------------------------------------------------------
// feat[b][d][p] = relu(sum_c enc_w[d][c] * xx[b][c][p])  (D=32)
// ---------------------------------------------------------------------------
__global__ __launch_bounds__(256) void encode_feat_k(const float* __restrict__ xx,
    const float* __restrict__ enc_w, float* __restrict__ feat) {
  __shared__ float ew[32 * 64];
  for (int i = threadIdx.x; i < 32 * 64; i += 256) ew[i] = enc_w[i];
  __syncthreads();
  int i = blockIdx.x * 256 + threadIdx.x;
  if (i >= 4 * HW3) return;
  int b = i / HW3, p = i - b * HW3;
  float z[32];
  #pragma unroll
  for (int d = 0; d < 32; ++d) z[d] = 0.f;
  for (int c = 0; c < 64; ++c) {
    float v = xx[((size_t)(b * 64 + c)) * HW3 + p];
    #pragma unroll
    for (int d = 0; d < 32; ++d) z[d] += ew[d * 64 + c] * v;
  }
  #pragma unroll
  for (int d = 0; d < 32; ++d) {
    float v = z[d];
    feat[((size_t)(b * 32 + d)) * HW3 + p] = v > 0.f ? v : 0.f;
  }
}

__global__ void zero_k(float* p, int n) {
  int i = blockIdx.x * 256 + threadIdx.x;
  if (i < n) p[i] = 0.f;
}

// ---------------------------------------------------------------------------
// Soft-VQ assignment + fused aggregation. The reference's view(1,D,-1) makes
// Zf[n][d] = featflat[d*NCOL + n] (feat buffer reinterpreted as (32, NCOL)).
// Accumulates S1[k][d] = sum_n a[n][k]*z[n][d] and Asum[k] via LDS partials.
// ---------------------------------------------------------------------------
__global__ __launch_bounds__(256) void soft_assign_k(const float* __restrict__ feat,
    const float* __restrict__ codewords, const float* __restrict__ scale,
    float* __restrict__ S1, float* __restrict__ Asum) {
  __shared__ float cw[48 * 32];
  __shared__ float cn[48];
  __shared__ float sc[48];
  __shared__ float Ep[48 * 32];
  __shared__ float Ap[48];
  int tid = threadIdx.x;
  for (int i = tid; i < 48 * 32; i += 256) { cw[i] = codewords[i]; Ep[i] = 0.f; }
  if (tid < 48) { Ap[tid] = 0.f; sc[tid] = scale[tid]; }
  __syncthreads();
  if (tid < 48) {
    float s = 0.f;
    for (int d = 0; d < 32; ++d) { float c = cw[tid * 32 + d]; s += c * c; }
    cn[tid] = s;
  }
  __syncthreads();
  int n = blockIdx.x * 256 + tid;
  if (n < NCOL) {
    float z[32]; float zz = 0.f;
    #pragma unroll
    for (int d = 0; d < 32; ++d) {
      float v = feat[(size_t)d * NCOL + n]; z[d] = v; zz += v * v;
    }
    float lg[48]; float mxl = -3.0e38f;
    for (int k = 0; k < 48; ++k) {
      float dot = 0.f;
      #pragma unroll
      for (int d = 0; d < 32; ++d) dot += z[d] * cw[k * 32 + d];
      float l = sc[k] * (zz + cn[k] - 2.f * dot);
      lg[k] = l; mxl = fmaxf(mxl, l);
    }
    float se = 0.f;
    for (int k = 0; k < 48; ++k) { float e = __expf(lg[k] - mxl); lg[k] = e; se += e; }
    float inv = 1.f / se;
    for (int k = 0; k < 48; ++k) {
      float a = lg[k] * inv;
      atomicAdd(&Ap[k], a);
      for (int d = 0; d < 32; ++d) atomicAdd(&Ep[k * 32 + d], a * z[d]);
    }
  }
  __syncthreads();
  for (int i = tid; i < 48 * 32; i += 256) atomicAdd(&S1[i], Ep[i]);
  if (tid < 48) atomicAdd(&Asum[tid], Ap[tid]);
}

// ---------------------------------------------------------------------------
// E = S1 - Asum*C; BatchNorm over D per k; E_sum = sum_k relu(En);
// gamma = sigmoid(E_sum @ attn_w.T + attn_b)   (64 values, shared over batch)
// ---------------------------------------------------------------------------
__global__ void bn_gamma_k(const float* __restrict__ S1, const float* __restrict__ Asum,
    const float* __restrict__ codewords, const float* __restrict__ bn_w,
    const float* __restrict__ bn_b, const float* __restrict__ attn_w,
    const float* __restrict__ attn_b, float* __restrict__ gamma) {
  __shared__ float En[48 * 32];
  __shared__ float Es[32];
  int tid = threadIdx.x;               // 64 threads
  if (tid < 48) {
    float e[32]; float mu = 0.f;
    float a = Asum[tid];
    for (int d = 0; d < 32; ++d) {
      float v = S1[tid * 32 + d] - a * codewords[tid * 32 + d];
      e[d] = v; mu += v;
    }
    mu *= (1.f / 32.f);
    float var = 0.f;
    for (int d = 0; d < 32; ++d) { float t = e[d] - mu; var += t * t; }
    var *= (1.f / 32.f);
    float inv = rsqrtf(var + 1e-5f);
    float w = bn_w[tid], bb = bn_b[tid];
    for (int d = 0; d < 32; ++d) {
      float v = (e[d] - mu) * inv * w + bb;
      En[tid * 32 + d] = v > 0.f ? v : 0.f;
    }
  }
  __syncthreads();
  if (tid < 32) {
    float s = 0.f;
    for (int k = 0; k < 48; ++k) s += En[k * 32 + tid];
    Es[tid] = s;
  }
  __syncthreads();
  float g = attn_b[tid];
  for (int d = 0; d < 32; ++d) g += Es[d] * attn_w[tid * 32 + d];
  gamma[tid] = 1.f / (1.f + __expf(-g));
}

// ---------------------------------------------------------------------------
// Fused: align-corners bilinear upsample of {c1,c2,xx*(1+gamma)} to 256x256,
// concat (192 ch) and 1x1 classifier (9 outputs), all in one pass.
// ---------------------------------------------------------------------------
__device__ __forceinline__ void upsample_accum(const float* __restrict__ src, int S,
    int b, int oh, int ow, int fbase, const float* __restrict__ W,
    const float* __restrict__ gmul, float acc[9]) {
  float scl = (float)(S - 1) * (1.0f / 255.0f);
  float fy = oh * scl; int y0 = (int)fy; if (y0 > S - 2) y0 = S - 2; float ty = fy - y0;
  float fx = ow * scl; int x0 = (int)fx; if (x0 > S - 2) x0 = S - 2; float tx = fx - x0;
  float w00 = (1.f - ty) * (1.f - tx), w01 = (1.f - ty) * tx;
  float w10 = ty * (1.f - tx),         w11 = ty * tx;
  const float* base = src + (size_t)b * 64 * S * S + y0 * S + x0;
  size_t cs = (size_t)S * S;
  for (int c = 0; c < 64; ++c) {
    const float* p = base + c * cs;
    float v = w00 * p[0] + w01 * p[1] + w10 * p[S] + w11 * p[S + 1];
    if (gmul) v *= gmul[c];
    const float* wr = W + (fbase + c);
    #pragma unroll
    for (int o = 0; o < 9; ++o) acc[o] += wr[o * 192] * v;
  }
}

__global__ __launch_bounds__(256) void final_cls_k(const float* __restrict__ c1,
    const float* __restrict__ c2, const float* __restrict__ xx,
    const float* __restrict__ gamma, const float* __restrict__ cls_w,
    const float* __restrict__ cls_b, float* __restrict__ out) {
  __shared__ float W[9 * 192];
  __shared__ float g1[64];
  for (int i = threadIdx.x; i < 9 * 192; i += 256) W[i] = cls_w[i];
  if (threadIdx.x < 64) g1[threadIdx.x] = 1.f + gamma[threadIdx.x];
  __syncthreads();
  int idx = blockIdx.x * 256 + threadIdx.x;
  if (idx >= 4 * 256 * 256) return;
  int b = idx >> 16, oh = (idx >> 8) & 255, ow = idx & 255;
  float acc[9];
  #pragma unroll
  for (int o = 0; o < 9; ++o) acc[o] = cls_b[o];
  upsample_accum(c1, 254, b, oh, ow, 0,   W, nullptr, acc);
  upsample_accum(c2, 250, b, oh, ow, 64,  W, nullptr, acc);
  upsample_accum(xx, 119, b, oh, ow, 128, W, g1,      acc);
  #pragma unroll
  for (int o = 0; o < 9; ++o)
    out[(((size_t)(b * 9 + o)) << 16) + oh * 256 + ow] = acc[o];
}

// ---------------------------------------------------------------------------
extern "C" void kernel_launch(void* const* d_in, const int* in_sizes, int n_in,
                              void* d_out, int out_size, void* d_ws, size_t ws_size,
                              hipStream_t stream) {
  (void)in_sizes; (void)n_in; (void)out_size; (void)ws_size;
  const float* x        = (const float*)d_in[0];
  const float* conv0_w  = (const float*)d_in[1];
  const float* conv0_b  = (const float*)d_in[2];
  const float* conv1_w  = (const float*)d_in[3];
  const float* conv1_b  = (const float*)d_in[4];
  const float* conv2_w  = (const float*)d_in[5];
  const float* conv2_b  = (const float*)d_in[6];
  const float* convp_w  = (const float*)d_in[7];
  const float* fc1_w    = (const float*)d_in[8];
  const float* fc2_w    = (const float*)d_in[9];
  const float* enc_w    = (const float*)d_in[10];
  const float* codewords= (const float*)d_in[11];
  const float* scale    = (const float*)d_in[12];
  const float* attn_w   = (const float*)d_in[13];
  const float* attn_b   = (const float*)d_in[14];
  const float* bn_w     = (const float*)d_in[15];
  const float* bn_b     = (const float*)d_in[16];
  const float* cls_w    = (const float*)d_in[17];
  const float* cls_b    = (const float*)d_in[18];

  char* ws = (char*)d_ws;
  size_t off = 0;
  auto carve = [&](size_t bytes) {
    char* p = ws + off;
    off += (bytes + 255) & ~(size_t)255;
    return p;
  };
  float* c1   = (float*)carve((size_t)4 * 64 * 254 * 254 * 4);   // 66.1 MB
  float* c2   = (float*)carve((size_t)4 * 64 * 250 * 250 * 4);   // 64.0 MB
  float* c3   = (float*)carve((size_t)4 * 64 * HW3 * 4);         // 14.5 MB
  float* xx   = (float*)carve((size_t)4 * 64 * HW3 * 4);         // 14.5 MB
  char*  rgn  = carve((size_t)4 * 64 * 125 * 125 * 4);           // 16.0 MB (pool, reused)
  float* pool = (float*)rgn;                                     // dead after conv2
  float* feat = (float*)rgn;                                     // 7.25 MB
  float* sp   = (float*)(rgn + 8 * 1024 * 1024);                 // 0.45 MB
  _Float16* wp0 = (_Float16*)carve((size_t)928 * 64 * 2);
  _Float16* wp1 = (_Float16*)carve((size_t)576 * 64 * 2);
  _Float16* wp2 = (_Float16*)carve((size_t)576 * 64 * 2);
  float* avg  = (float*)carve(256 * 4);
  float* mxv  = (float*)carve(256 * 4);
  float* gate = (float*)carve(256 * 4);
  float* S1   = (float*)carve((48 * 32 + 48) * 4);               // S1 then Asum
  float* Asum = S1 + 48 * 32;
  float* gamma= (float*)carve(64 * 4);

  // Pack conv weights to fp16 n-major (K padded to multiple of 32).
  pack_weights_k<<<(928 * 64 + 255) / 256, 256, 0, stream>>>(conv0_w, wp0, 927, 928);
  pack_weights_k<<<(576 * 64 + 255) / 256, 256, 0, stream>>>(conv1_w, wp1, 576, 576);
  pack_weights_k<<<(576 * 64 + 255) / 256, 256, 0, stream>>>(conv2_w, wp2, 576, 576);

  // conv0: 103->64, dil 1, 256 -> 254
  { int M = 4 * 254 * 254;
    conv3x3_wmma_k<<<(M + 63) / 64, 256, 0, stream>>>(x, wp0, conv0_b, c1,
        4, 103, 256, 256, 254, 254, 1, 927, 29); }
  // conv1: 64->64, dil 2, 254 -> 250
  { int M = 4 * 250 * 250;
    conv3x3_wmma_k<<<(M + 63) / 64, 256, 0, stream>>>(c1, wp1, conv1_b, c2,
        4, 64, 254, 254, 250, 250, 2, 576, 18); }
  avgpool2_k<<<(4 * 64 * 125 * 125 + 255) / 256, 256, 0, stream>>>(c2, pool);
  // conv2: 64->64, dil 3, 125 -> 119
  { int M = 4 * 119 * 119;
    conv3x3_wmma_k<<<(M + 63) / 64, 256, 0, stream>>>(pool, wp2, conv2_b, c3,
        4, 64, 125, 125, 119, 119, 3, 576, 18); }

  chan_reduce_k<<<256, 256, 0, stream>>>(c3, avg, mxv);
  gate_k<<<1, 256, 0, stream>>>(avg, mxv, fc1_w, fc2_w, gate);
  spatial_pool_k<<<(4 * HW3 + 255) / 256, 256, 0, stream>>>(c3, gate, sp);
  spatial_attn_k<<<(4 * HW3 + 255) / 256, 256, 0, stream>>>(sp, convp_w, c3, xx);
  encode_feat_k<<<(4 * HW3 + 255) / 256, 256, 0, stream>>>(xx, enc_w, feat);
  zero_k<<<(48 * 32 + 48 + 255) / 256, 256, 0, stream>>>(S1, 48 * 32 + 48);
  soft_assign_k<<<(NCOL + 255) / 256, 256, 0, stream>>>(feat, codewords, scale, S1, Asum);
  bn_gamma_k<<<1, 64, 0, stream>>>(S1, Asum, codewords, bn_w, bn_b, attn_w, attn_b, gamma);
  final_cls_k<<<(4 * 256 * 256) / 256, 256, 0, stream>>>(c1, c2, xx, gamma,
      cls_w, cls_b, (float*)d_out);
}
